// Batch_Contrastive_18116172055214
// MI455X (gfx1250) — matrix operand order repaired and checked by reading, not compiled, not run on hardware
//
#include <hip/hip_runtime.h>
#include <hip/hip_bf16.h>

typedef __attribute__((ext_vector_type(16))) __bf16 v16bf;
typedef __attribute__((ext_vector_type(8)))  __bf16 v8bf;
typedef __attribute__((ext_vector_type(8)))  float  v8f;

#define N_TOT 8192
#define C_DIM 256
#define HW    1024
#define B_DIM 8
#define NPART 16          // column partitions per row tile
#define COLS_PER_PART (N_TOT / NPART)   // 512

// ---------------------------------------------------------------------------
// [b, c, h, w] f32  ->  [n, c] bf16 with n = b*hw + (y*w+x)
// Reads fully coalesced along hw; writes are scattered bf16 (tiny traffic).
__global__ void convert_bchw_to_nc_bf16(const float* __restrict__ src,
                                        __bf16* __restrict__ dst) {
    int t = blockIdx.x * blockDim.x + threadIdx.x;   // 0 .. 8*256*1024-1
    int p = t & (HW - 1);
    int k = (t >> 10) & (C_DIM - 1);
    int b = t >> 18;                                  // 256*1024 = 2^18
    dst[(size_t)(b * HW + p) * C_DIM + k] = (__bf16)src[t];
}

// [b, 3, h, w] f32 -> [n, 4] f32 (padded float4 rows, .w = 0)
__global__ void convert_color(const float* __restrict__ src,
                              float* __restrict__ dst4) {
    int t = blockIdx.x * blockDim.x + threadIdx.x;   // 0 .. 8192*4-1
    int c = t & 3;
    int n = t >> 2;
    int b = n >> 10;
    int p = n & (HW - 1);
    dst4[t] = (c < 3) ? src[(b * 3 + c) * HW + p] : 0.0f;
}

// ---------------------------------------------------------------------------
// Flash-attention style: each wave = one 16-row tile x one 512-column slice.
// Writes per-row partial (m, l, s_batch, c0, c1, c2) for the LSE merge.
__launch_bounds__(256)
__global__ void flash_attn_color(const __bf16* __restrict__ Q,
                                 const __bf16* __restrict__ K,
                                 const float4* __restrict__ FC,
                                 float* __restrict__ partial) {
    const int lane = threadIdx.x & 31;
    const int l16  = lane & 15;
    const int g    = lane >> 4;
    const int wave = (blockIdx.x * blockDim.x + threadIdx.x) >> 5;
    const int row_tile = wave >> 4;          // 0..511
    const int part     = wave & (NPART - 1); // 0..15
    const int row_base = row_tile * 16;
    const int row_batch = row_base >> 10;

    // ---- A fragments: 16 rows x K=256, contiguous k = 16g + s per lane half ----
    const __bf16* qrow = Q + (size_t)(row_base + l16) * C_DIM;
    v16bf aQ[8];
#pragma unroll
    for (int kt = 0; kt < 8; ++kt) {
        union { v16bf v; v8bf h[2]; } u;
        u.h[0] = *reinterpret_cast<const v8bf*>(qrow + kt * 32 + g * 16);
        u.h[1] = *reinterpret_cast<const v8bf*>(qrow + kt * 32 + g * 16 + 8);
        aQ[kt] = u.v;
    }

    float m_run[8], l_run[8], s_run[8], c0a[8], c1a[8], c2a[8];
#pragma unroll
    for (int v = 0; v < 8; ++v) {
        m_run[v] = -3.0e38f;
        l_run[v] = 0.f; s_run[v] = 0.f;
        c0a[v] = 0.f; c1a[v] = 0.f; c2a[v] = 0.f;
    }

    const int col_begin = part * COLS_PER_PART;
    const int col_end   = col_begin + COLS_PER_PART;

    for (int cb = col_begin; cb < col_end; cb += 32) {
        const bool same_batch = ((cb >> 10) == row_batch);

        v8f acc0 = {0.f,0.f,0.f,0.f,0.f,0.f,0.f,0.f};
        v8f acc1 = {0.f,0.f,0.f,0.f,0.f,0.f,0.f,0.f};
        // B fragments: lane holds key-row (col+l16), contiguous K span
        const __bf16* kr0 = K + (size_t)(cb + l16) * C_DIM;
        const __bf16* kr1 = kr0 + 16 * C_DIM;
#pragma unroll
        for (int kt = 0; kt < 8; ++kt) {
            union { v16bf v; v8bf h[2]; } b0, b1;
            b0.h[0] = *reinterpret_cast<const v8bf*>(kr0 + kt * 32 + g * 16);
            b0.h[1] = *reinterpret_cast<const v8bf*>(kr0 + kt * 32 + g * 16 + 8);
            b1.h[0] = *reinterpret_cast<const v8bf*>(kr1 + kt * 32 + g * 16);
            b1.h[1] = *reinterpret_cast<const v8bf*>(kr1 + kt * 32 + g * 16 + 8);
            acc0 = __builtin_amdgcn_wmma_f32_16x16x32_bf16(
                       false, aQ[kt], false, b0.v, (short)0, acc0, false, false);
            acc1 = __builtin_amdgcn_wmma_f32_16x16x32_bf16(
                       false, aQ[kt], false, b1.v, (short)0, acc1, false, false);
        }

        float4 fc0 = FC[cb + l16];
        float4 fc1 = FC[cb + 16 + l16];

#pragma unroll
        for (int v = 0; v < 8; ++v) {
            float x0 = acc0[v], x1 = acc1[v];
            // row max across the 16-lane group (row v+8g)
            float t = fmaxf(x0, x1);
#pragma unroll
            for (int off = 1; off < 16; off <<= 1)
                t = fmaxf(t, __shfl_xor(t, off, 16));
            float newm  = fmaxf(m_run[v], t);
            float scale = __expf(m_run[v] - newm);
            m_run[v] = newm;
            float p0 = __expf(x0 - newm);
            float p1 = __expf(x1 - newm);
            float ps = p0 + p1;
            l_run[v] = fmaf(l_run[v], scale, ps);
            s_run[v] = fmaf(s_run[v], scale, same_batch ? ps : 0.0f);
            c0a[v] = fmaf(c0a[v], scale, fmaf(p0, fc0.x, p1 * fc1.x));
            c1a[v] = fmaf(c1a[v], scale, fmaf(p0, fc0.y, p1 * fc1.y));
            c2a[v] = fmaf(c2a[v], scale, fmaf(p0, fc0.z, p1 * fc1.z));
        }
    }

    // ---- reduce lane-partials across the 16-lane group, emit partials ----
#pragma unroll
    for (int v = 0; v < 8; ++v) {
        float l = l_run[v], s = s_run[v];
        float r0 = c0a[v], r1 = c1a[v], r2 = c2a[v];
#pragma unroll
        for (int off = 1; off < 16; off <<= 1) {
            l  += __shfl_xor(l,  off, 16);
            s  += __shfl_xor(s,  off, 16);
            r0 += __shfl_xor(r0, off, 16);
            r1 += __shfl_xor(r1, off, 16);
            r2 += __shfl_xor(r2, off, 16);
        }
        if (l16 == 0) {
            int row = row_base + v + 8 * g;
            float* dst = partial + ((size_t)row * NPART + part) * 8;
            dst[0] = m_run[v]; dst[1] = l; dst[2] = s;
            dst[3] = r0; dst[4] = r1; dst[5] = r2;
        }
    }
}

// ---------------------------------------------------------------------------
// Deterministic log-sum-exp merge over the 16 column partitions of each row.
__global__ void merge_rows(const float* __restrict__ partial,
                           float* __restrict__ out_color,
                           float* __restrict__ rowloss) {
    int row = blockIdx.x * blockDim.x + threadIdx.x;
    if (row >= N_TOT) return;
    const float* pr = partial + (size_t)row * NPART * 8;
    float mstar = -3.0e38f;
#pragma unroll
    for (int p = 0; p < NPART; ++p) mstar = fmaxf(mstar, pr[p * 8 + 0]);
    float l = 0.f, s = 0.f, r0 = 0.f, r1 = 0.f, r2 = 0.f;
#pragma unroll
    for (int p = 0; p < NPART; ++p) {
        float w = __expf(pr[p * 8 + 0] - mstar);
        l  = fmaf(pr[p * 8 + 1], w, l);
        s  = fmaf(pr[p * 8 + 2], w, s);
        r0 = fmaf(pr[p * 8 + 3], w, r0);
        r1 = fmaf(pr[p * 8 + 4], w, r1);
        r2 = fmaf(pr[p * 8 + 5], w, r2);
    }
    float inv = 1.0f / l;
    int b = row >> 10, pix = row & (HW - 1);
    out_color[(b * 3 + 0) * HW + pix] = r0 * inv;
    out_color[(b * 3 + 1) * HW + pix] = r1 * inv;
    out_color[(b * 3 + 2) * HW + pix] = r2 * inv;
    rowloss[row] = fabsf(s * inv - 1.0f);
}

// Fixed-tree reduction -> deterministic scalar loss.
__global__ void loss_reduce(const float* __restrict__ rowloss,
                            float* __restrict__ out_loss) {
    __shared__ float sm[256];
    float a = 0.0f;
    for (int i = threadIdx.x; i < N_TOT; i += 256) a += rowloss[i];
    sm[threadIdx.x] = a;
    __syncthreads();
    for (int s = 128; s > 0; s >>= 1) {
        if (threadIdx.x < s) sm[threadIdx.x] += sm[threadIdx.x + s];
        __syncthreads();
    }
    if (threadIdx.x == 0) *out_loss = sm[0] * (1.0f / N_TOT);
}

// ---------------------------------------------------------------------------
extern "C" void kernel_launch(void* const* d_in, const int* in_sizes, int n_in,
                              void* d_out, int out_size, void* d_ws, size_t ws_size,
                              hipStream_t stream) {
    const float* feat2  = (const float*)d_in[0];  // queries (rows of affinity)
    const float* feat1  = (const float*)d_in[1];  // keys    (cols of affinity)
    const float* fcolor = (const float*)d_in[2];
    float* out = (float*)d_out;                   // 24576 colors + 1 loss

    char* ws = (char*)d_ws;
    __bf16* Qbf    = (__bf16*)ws;                                  // 4 MB
    __bf16* Kbf    = (__bf16*)(ws + (size_t)4 * 1024 * 1024);      // 4 MB
    float*  fc4    = (float*)(ws + (size_t)8 * 1024 * 1024);       // 128 KB in 1 MB slot
    float*  part   = (float*)(ws + (size_t)9 * 1024 * 1024);       // 4 MB partials
    float*  rowls  = (float*)(ws + (size_t)13 * 1024 * 1024);      // 32 KB

    const int nelem = B_DIM * C_DIM * HW;                          // 2^21
    convert_bchw_to_nc_bf16<<<nelem / 256, 256, 0, stream>>>(feat2, Qbf);
    convert_bchw_to_nc_bf16<<<nelem / 256, 256, 0, stream>>>(feat1, Kbf);
    convert_color<<<(N_TOT * 4) / 256, 256, 0, stream>>>(fcolor, fc4);

    // 512 row tiles x 16 column partitions = 8192 waves = 1024 blocks x 8 waves
    flash_attn_color<<<(N_TOT / 16) * NPART * 32 / 256, 256, 0, stream>>>(
        Qbf, Kbf, (const float4*)fc4, part);

    merge_rows<<<N_TOT / 256, 256, 0, stream>>>(part, out, rowls);
    loss_reduce<<<1, 256, 0, stream>>>(rowls, out + 24576);
}
